// PDDFormer_60069412602097
// MI455X (gfx1250) — matrix operand order repaired and checked by reading, not compiled, not run on hardware
//
#include <hip/hip_runtime.h>
#include <hip/hip_bf16.h>

// ---------- types ----------
typedef unsigned short u16;
typedef __attribute__((ext_vector_type(16))) __bf16 v16bf;
typedef __attribute__((ext_vector_type(8)))  float  v8f;
typedef __attribute__((ext_vector_type(4)))  unsigned int uint4v;
union Frag16 { v16bf bf; uint4v u[2]; };

#define kN 8192
#define kE 131072
#define kC 256
#define kB 64

__device__ __forceinline__ u16 f2b(float f) {
    unsigned u = __float_as_uint(f);
    unsigned r = (u + 0x7FFFu + ((u >> 16) & 1u)) >> 16;
    return (u16)r;
}
__device__ __forceinline__ float siluf(float v) { return v * (1.f / (1.f + __expf(-v))); }

// ---- CDNA5 async global->LDS copy (16B per lane, tracked by ASYNCcnt) ----
__device__ __forceinline__ void async_b128(unsigned ldsOff, const void* gaddr) {
    asm volatile("global_load_async_to_lds_b128 %0, %1, off"
                 :: "v"(ldsOff), "v"(gaddr) : "memory");
}
__device__ __forceinline__ void wait_async0() {
    asm volatile("s_wait_asynccnt 0x0" ::: "memory");
}

// ---------- generic WMMA GEMM: out = act(A[MxK]bf16 @ Bt[NcxK]bf16^T + bias [+resid]) ----------
// 256 threads = 8 waves; block tile 128 rows x 64 cols; B tile double-buffered in LDS,
// filled by async global->LDS loads that overlap the WMMAs of the previous K-chunk.
__global__ __launch_bounds__(256) void k_gemm(
    const u16* __restrict__ A, const u16* __restrict__ Bt,
    const float* __restrict__ bias, const float* __restrict__ resid,
    float* __restrict__ outF, u16* __restrict__ outB,
    int M, int K, int Nc, int act)
{
    __shared__ u16 ldsB[2][64 * 32];
    const int tid  = threadIdx.x;
    const int wave = tid >> 5, lane = tid & 31;
    const int hi = lane >> 4, l16 = lane & 15;
    const int rowBase = blockIdx.x * 128 + wave * 16;
    const int colBase = blockIdx.y * 64;
    // staging assignment: thread -> (column row sn, 8-elem quarter sq); 256 x 16B = 4KB tile
    const int sn = tid >> 2, sq = tid & 3;
    const u16* sB = Bt + (size_t)(colBase + sn) * K + sq * 8;
    const unsigned ldsOff0 = (unsigned)(uintptr_t)&ldsB[0][sn * 32 + sq * 8];
    const unsigned ldsOff1 = (unsigned)(uintptr_t)&ldsB[1][sn * 32 + sq * 8];

    v8f acc[4] = {};
    const u16* aPtr = A + (size_t)(rowBase + l16) * K;
    const int nch = K >> 5;

    async_b128(ldsOff0, sB);                       // prologue: chunk 0 -> buf 0
    for (int i = 0; i < nch; ++i) {
        const int cur = i & 1;
        wait_async0();                             // my chunk-i stores to LDS done
        __syncthreads();                           // everyone's done; prior reads of buf^1 done
        if (i + 1 < nch)                           // overlap: chunk i+1 -> other buffer
            async_b128(cur ? ldsOff0 : ldsOff1, sB + (size_t)(i + 1) * 32);
        const int kk = i * 32;
        Frag16 fa;
        {   // A layout: lane half selects kb in {0,8}; elems [kb,kb+8) and [kb+16,kb+24)
            const u16* p = aPtr + kk + hi * 8;
            fa.u[0] = *(const uint4v*)p;
            fa.u[1] = *(const uint4v*)(p + 16);
        }
        if (i + 1 < nch) __builtin_prefetch(aPtr + kk + 32 + hi * 8, 0, 0);
        const u16* bbuf = &ldsB[cur][0];
#pragma unroll
        for (int j = 0; j < 4; ++j) {
            Frag16 fb;  // B layout: lane = column, lane half selects kb in {0,8}; elems [kb,kb+8) and [kb+16,kb+24)
            const u16* p = bbuf + (j * 16 + l16) * 32 + hi * 8;
            fb.u[0] = *(const uint4v*)p;
            fb.u[1] = *(const uint4v*)(p + 16);
            acc[j] = __builtin_amdgcn_wmma_f32_16x16x32_bf16(
                false, fa.bf, false, fb.bf, (short)0, acc[j], false, false);
        }
    }
#pragma unroll
    for (int j = 0; j < 4; ++j)
#pragma unroll
        for (int i = 0; i < 8; ++i) {
            int m = rowBase + hi * 8 + i;
            int n = colBase + j * 16 + l16;
            float v = acc[j][i] + bias[n];
            if (resid) v += resid[(size_t)m * Nc + n];
            if (act == 1) v = siluf(v);
            if (outF) outF[(size_t)m * Nc + n] = v;
            if (outB) outB[(size_t)m * Nc + n] = f2b(v);
        }
}

// ---------- gathered concat GEMM: A row = [k[dst], k[src], e], K=768 ----------
__global__ __launch_bounds__(256) void k_gemm_g3(
    const u16* __restrict__ Kb, const u16* __restrict__ Eb,
    const int* __restrict__ ei, const u16* __restrict__ Bt,
    const float* __restrict__ bias, float* __restrict__ outF, int Ecnt)
{
    __shared__ u16 ldsB[2][64 * 32];
    const int tid  = threadIdx.x;
    const int wave = tid >> 5, lane = tid & 31;
    const int hi = lane >> 4, l16 = lane & 15;
    const int rowBase = blockIdx.x * 128 + wave * 16;
    const int colBase = blockIdx.y * 64;
    const int sn = tid >> 2, sq = tid & 3;
    const u16* sB = Bt + (size_t)(colBase + sn) * 768 + sq * 8;
    const unsigned ldsOff0 = (unsigned)(uintptr_t)&ldsB[0][sn * 32 + sq * 8];
    const unsigned ldsOff1 = (unsigned)(uintptr_t)&ldsB[1][sn * 32 + sq * 8];

    const int m = rowBase + l16;
    const int idst = ei[Ecnt + m], isrc = ei[m];
    const u16* bases[3] = { Kb + (size_t)idst * kC, Kb + (size_t)isrc * kC,
                            Eb + (size_t)m * kC };
    v8f acc[4] = {};
    const int nch = 24;                             // 768 / 32

    async_b128(ldsOff0, sB);
    for (int i = 0; i < nch; ++i) {
        const int cur = i & 1;
        wait_async0();
        __syncthreads();
        if (i + 1 < nch)
            async_b128(cur ? ldsOff0 : ldsOff1, sB + (size_t)(i + 1) * 32);
        const int seg = i >> 3, kc = (i & 7) << 5;
        Frag16 fa;
        {
            const u16* p = bases[seg] + kc + hi * 8;
            fa.u[0] = *(const uint4v*)p;
            fa.u[1] = *(const uint4v*)(p + 16);
        }
        if ((i & 7) != 7) __builtin_prefetch(bases[seg] + kc + 32 + hi * 8, 0, 0);
        const u16* bbuf = &ldsB[cur][0];
#pragma unroll
        for (int j = 0; j < 4; ++j) {
            Frag16 fb;
            const u16* p = bbuf + (j * 16 + l16) * 32 + hi * 8;
            fb.u[0] = *(const uint4v*)p;
            fb.u[1] = *(const uint4v*)(p + 16);
            acc[j] = __builtin_amdgcn_wmma_f32_16x16x32_bf16(
                false, fa.bf, false, fb.bf, (short)0, acc[j], false, false);
        }
    }
#pragma unroll
    for (int j = 0; j < 4; ++j)
#pragma unroll
        for (int i = 0; i < 8; ++i) {
            int mr = rowBase + hi * 8 + i;
            int n = colBase + j * 16 + l16;
            outF[(size_t)mr * kC + n] = acc[j][i] + bias[n];
        }
}

// ---------- small conversion / prep kernels ----------
__global__ __launch_bounds__(256) void k_cvt_transpose(
    const float* __restrict__ W, u16* __restrict__ Wt, int K, int Nn, int Kp, int total)
{
    int idx = blockIdx.x * 256 + threadIdx.x;
    if (idx >= total) return;
    int n = idx / Kp, kx = idx % Kp;
    Wt[idx] = (kx < K) ? f2b(W[(size_t)kx * Nn + n]) : (u16)0;
}

__global__ __launch_bounds__(256) void k_pad_cvt(
    const float* __restrict__ A, u16* __restrict__ Ab, int K, int Kp, int total)
{
    int idx = blockIdx.x * 256 + threadIdx.x;
    if (idx >= total) return;
    int r = idx / Kp, kx = idx % Kp;
    Ab[idx] = (kx < K) ? f2b(A[(size_t)r * K + kx]) : (u16)0;
}

__global__ __launch_bounds__(256) void k_rbf(const float* __restrict__ ea,
                                             u16* __restrict__ rbfb)
{
    int r = blockIdx.x, c = threadIdx.x;
    float x0 = ea[r * 3], x1 = ea[r * 3 + 1], x2 = ea[r * 3 + 2];
    float d = -rsqrtf(x0 * x0 + x1 * x1 + x2 * x2);
    float center = -6.f + 6.f * (float)c / 255.f;
    float t = d - center;
    rbfb[(size_t)r * kC + c] = f2b(__expf(-(255.f / 6.f) * t * t));
}

// ---------- attention gate: msgB = bf16(msgRaw * sigmoid(LN(q[dst]*keyE*scale))) ----------
__global__ __launch_bounds__(256) void k_alpha_gate(
    const float* __restrict__ q, const float* __restrict__ keyE,
    const float* __restrict__ msgRaw, const int* __restrict__ ei,
    const float* __restrict__ g, const float* __restrict__ bb,
    u16* __restrict__ msgB)
{
    __shared__ float s1[256], s2[256];
    int r = blockIdx.x, c = threadIdx.x;
    int dst = ei[kE + r];
    float a = q[(size_t)dst * kC + c] * keyE[(size_t)r * kC + c] * 0.0625f;
    s1[c] = a; s2[c] = a * a; __syncthreads();
    for (int s = 128; s > 0; s >>= 1) {
        if (c < s) { s1[c] += s1[c + s]; s2[c] += s2[c + s]; }
        __syncthreads();
    }
    float mean = s1[0] * (1.f / 256.f);
    float var = s2[0] * (1.f / 256.f) - mean * mean;
    float ln = (a - mean) * rsqrtf(var + 1e-5f) * g[c] + bb[c];
    float gate = 1.f / (1.f + __expf(-ln));
    msgB[(size_t)r * kC + c] = f2b(msgRaw[(size_t)r * kC + c] * gate);
}

// ---------- row LN then segment-sum scatter into agg[dst] ----------
__global__ __launch_bounds__(256) void k_ln_scatter(
    const float* __restrict__ msg2, const int* __restrict__ ei,
    const float* __restrict__ g, const float* __restrict__ bb,
    float* __restrict__ agg)
{
    __shared__ float s1[256], s2[256];
    int r = blockIdx.x, c = threadIdx.x;
    float a = msg2[(size_t)r * kC + c];
    s1[c] = a; s2[c] = a * a; __syncthreads();
    for (int s = 128; s > 0; s >>= 1) {
        if (c < s) { s1[c] += s1[c + s]; s2[c] += s2[c + s]; }
        __syncthreads();
    }
    float mean = s1[0] * (1.f / 256.f);
    float var = s2[0] * (1.f / 256.f) - mean * mean;
    float ln = (a - mean) * rsqrtf(var + 1e-5f) * g[c] + bb[c];
    int dst = ei[kE + r];
    atomicAdd(&agg[(size_t)dst * kC + c], ln);
}

// ---------- column sum / sumsq (256 rows per block) ----------
__global__ __launch_bounds__(256) void k_colstats(const float* __restrict__ X,
                                                  float* __restrict__ s,
                                                  float* __restrict__ sq)
{
    int c = threadIdx.x;
    size_t r0 = (size_t)blockIdx.x * 256;
    float a = 0.f, a2 = 0.f;
    for (int rr = 0; rr < 256; ++rr) {
        float v = X[(r0 + rr) * kC + c];
        a += v; a2 += v * v;
    }
    atomicAdd(&s[c], a);
    atomicAdd(&sq[c], a2);
}

__global__ __launch_bounds__(256) void k_bn_softplus(
    const float* __restrict__ xold, const float* __restrict__ agg,
    const float* __restrict__ s, const float* __restrict__ sq,
    const float* __restrict__ g, const float* __restrict__ bb,
    float* __restrict__ xnew, u16* __restrict__ xb)
{
    int idx = blockIdx.x * 256 + threadIdx.x;
    int c = idx & 255;
    float m = s[c] * (1.f / (float)kN);
    float v = sq[c] * (1.f / (float)kN) - m * m;
    float t = (agg[idx] - m) * rsqrtf(v + 1e-5f) * g[c] + bb[c];
    float z = xold[idx] + t;
    float sp = (z > 20.f) ? z : log1pf(__expf(z));
    xnew[idx] = sp;
    xb[idx] = f2b(sp);
}

__global__ __launch_bounds__(256) void k_add(float* __restrict__ adj,
                                             const float* __restrict__ x)
{
    int idx = blockIdx.x * 256 + threadIdx.x;
    adj[idx] += x[idx];
}

__global__ __launch_bounds__(256) void k_bn_cvt(
    const float* __restrict__ adj, const float* __restrict__ s,
    const float* __restrict__ sq, const float* __restrict__ g,
    const float* __restrict__ bb, u16* __restrict__ outb)
{
    int idx = blockIdx.x * 256 + threadIdx.x;
    int c = idx & 255;
    float m = s[c] * (1.f / (float)kN);
    float v = sq[c] * (1.f / (float)kN) - m * m;
    outb[idx] = f2b((adj[idx] - m) * rsqrtf(v + 1e-5f) * g[c] + bb[c]);
}

__global__ __launch_bounds__(256) void k_split(const float* __restrict__ a2,
                                               u16* __restrict__ x1b,
                                               float* __restrict__ x2f)
{
    int idx = blockIdx.x * 256 + threadIdx.x;
    int r = idx >> 8, c = idx & 255;
    float v1 = a2[(size_t)r * 512 + c];
    float v2 = a2[(size_t)r * 512 + 256 + c];
    x1b[idx] = f2b(v1);
    x2f[idx] = 0.5f * v2 * (1.f + erff(v2 * 0.70710678f));
}

__global__ __launch_bounds__(256) void k_mul_cvt(const float* __restrict__ a,
                                                 const float* __restrict__ b,
                                                 u16* __restrict__ o)
{
    int idx = blockIdx.x * 256 + threadIdx.x;
    o[idx] = f2b(a[idx] * b[idx]);
}

// ---------- pooling + head ----------
__global__ __launch_bounds__(256) void k_count(const int* __restrict__ batch,
                                               float* __restrict__ counts)
{
    int r = blockIdx.x * 256 + threadIdx.x;
    atomicAdd(&counts[batch[r]], 1.f);
}
__global__ __launch_bounds__(256) void k_pool(const float* __restrict__ x,
                                              const int* __restrict__ batch,
                                              float* __restrict__ pooled)
{
    int idx = blockIdx.x * 256 + threadIdx.x;
    int r = idx >> 8, c = idx & 255;
    atomicAdd(&pooled[(size_t)batch[r] * kC + c], x[idx]);
}
__global__ __launch_bounds__(256) void k_poolfin(float* __restrict__ pooled,
                                                 const float* __restrict__ counts)
{
    int idx = blockIdx.x * 256 + threadIdx.x;
    pooled[idx] /= fmaxf(counts[idx >> 8], 1.f);
}
__global__ __launch_bounds__(256) void k_head1(const float* __restrict__ pm,
                                               const float* __restrict__ fcW,
                                               const float* __restrict__ fcb,
                                               float* __restrict__ cf)
{
    int idx = blockIdx.x * 256 + threadIdx.x;
    int b = idx >> 8, c = idx & 255;
    float acc = fcb[c];
    const float* row = pm + (size_t)b * kC;
    for (int k2 = 0; k2 < 256; ++k2) acc += row[k2] * fcW[(size_t)k2 * kC + c];
    cf[idx] = pm[idx] + siluf(acc);
}
__global__ __launch_bounds__(256) void k_head2(const float* __restrict__ cf,
                                               const float* __restrict__ oW,
                                               const float* __restrict__ ob,
                                               float* __restrict__ out)
{
    __shared__ float s[256];
    int b = blockIdx.x, c = threadIdx.x;
    s[c] = cf[(size_t)b * kC + c] * oW[c];
    __syncthreads();
    for (int t = 128; t > 0; t >>= 1) {
        if (c < t) s[c] += s[c + t];
        __syncthreads();
    }
    if (c == 0) out[b] = s[0] + ob[0];
}

// ============================ host orchestration ============================
extern "C" void kernel_launch(void* const* d_in, const int* in_sizes, int n_in,
                              void* d_out, int out_size, void* d_ws, size_t ws_size,
                              hipStream_t stream)
{
    (void)in_sizes; (void)n_in; (void)out_size; (void)ws_size;
    const float* node     = (const float*)d_in[0];
    const int*   ei       = (const int*)  d_in[1];
    const float* edge_attr= (const float*)d_in[2];
    const float* pdd      = (const float*)d_in[3];
    const int*   batch    = (const int*)  d_in[4];
    const float* atom_W1  = (const float*)d_in[5];
    const float* atom_b1  = (const float*)d_in[6];
    const float* atom_W2  = (const float*)d_in[7];
    const float* atom_b2  = (const float*)d_in[8];
    const float* edge_W   = (const float*)d_in[9];
    const float* edge_b   = (const float*)d_in[10];
    const float* pdd_W    = (const float*)d_in[11];
    const float* pdd_b    = (const float*)d_in[12];
    const float* Wq = (const float*)d_in[13]; const float* bq = (const float*)d_in[14];
    const float* Wk = (const float*)d_in[15]; const float* bk = (const float*)d_in[16];
    const float* Wv = (const float*)d_in[17]; const float* bv = (const float*)d_in[18];
    const float* We = (const float*)d_in[19]; const float* be = (const float*)d_in[20];
    const float* Wku= (const float*)d_in[21]; const float* bku= (const float*)d_in[22];
    const float* Wmu= (const float*)d_in[23]; const float* bmu= (const float*)d_in[24];
    const float* Wm = (const float*)d_in[25]; const float* bm = (const float*)d_in[26];
    const float* ln_m_g = (const float*)d_in[27]; const float* ln_m_b = (const float*)d_in[28];
    const float* ln_a_g = (const float*)d_in[29]; const float* ln_a_b = (const float*)d_in[30];
    const float* bn_g   = (const float*)d_in[31]; const float* bn_b   = (const float*)d_in[32];
    const float* pW1 = (const float*)d_in[33]; const float* pb1 = (const float*)d_in[34];
    const float* pW2 = (const float*)d_in[35]; const float* pb2 = (const float*)d_in[36];
    const float* pW3 = (const float*)d_in[37]; const float* pb3 = (const float*)d_in[38];
    const float* pbn_g = (const float*)d_in[39]; const float* pbn_b = (const float*)d_in[40];
    const float* fc_W = (const float*)d_in[41]; const float* fc_b = (const float*)d_in[42];
    const float* out_W= (const float*)d_in[43]; const float* out_b= (const float*)d_in[44];
    float* outp = (float*)d_out;

    // ---- workspace carve-up ----
    char* base = (char*)d_ws;
    size_t off = 0;
    auto alloc = [&](size_t bytes) -> char* {
        char* p = base + off;
        off += (bytes + 255) & ~(size_t)255;
        return p;
    };
    u16* tW_a1  = (u16*)alloc(256 * 96 * 2);
    u16* tW_a2  = (u16*)alloc(256 * 256 * 2);
    u16* tW_ed  = (u16*)alloc(256 * 256 * 2);
    u16* tW_pdd = (u16*)alloc(256 * 64 * 2);
    u16* tW_q   = (u16*)alloc(3 * 256 * 256 * 2);
    u16* tW_k   = (u16*)alloc(3 * 256 * 256 * 2);
    u16* tW_v   = (u16*)alloc(3 * 256 * 256 * 2);
    u16* tW_e   = (u16*)alloc(3 * 256 * 256 * 2);
    u16* tW_m   = (u16*)alloc(3 * 256 * 256 * 2);
    u16* tW_ku  = (u16*)alloc(3 * 256 * 768 * 2);
    u16* tW_mu  = (u16*)alloc(3 * 256 * 768 * 2);
    u16* tW_p1  = (u16*)alloc(2 * 512 * 256 * 2);
    u16* tW_p2  = (u16*)alloc(2 * 256 * 256 * 2);
    u16* tW_p3  = (u16*)alloc(2 * 256 * 256 * 2);
    u16* nodeb  = (u16*)alloc((size_t)kN * 96 * 2);
    u16* pddb   = (u16*)alloc((size_t)kN * 64 * 2);
    u16* h1b    = (u16*)alloc((size_t)kN * kC * 2);
    float* xf   = (float*)alloc((size_t)kN * kC * 4);
    u16*   xb   = (u16*)  alloc((size_t)kN * kC * 2);
    float* pf   = (float*)alloc((size_t)kN * kC * 4);
    float* qf   = (float*)alloc((size_t)kN * kC * 4);
    u16*   kbuf = (u16*)  alloc((size_t)kN * kC * 2);
    u16*   vbuf = (u16*)  alloc((size_t)kN * kC * 2);
    u16*   einb = (u16*)  alloc((size_t)kE * kC * 2);
    u16*   ebuf = (u16*)  alloc((size_t)kE * kC * 2); // rbf -> e -> gated msg (aliased phases)
    float* keyE = (float*)alloc((size_t)kE * kC * 4); // keyE -> msg2 (aliased phases)
    float* msgR = (float*)alloc((size_t)kE * kC * 4);
    float* agg  = (float*)alloc((size_t)kN * kC * 4);
    float* cs   = (float*)alloc(kC * 4);
    float* csq  = (float*)alloc(kC * 4);
    float* a2   = (float*)alloc((size_t)kN * 512 * 4);
    float* x2f  = (float*)alloc((size_t)kN * kC * 4);
    u16*   x1b  = (u16*)  alloc((size_t)kN * kC * 2);
    float* x1o  = (float*)alloc((size_t)kN * kC * 4);
    u16*   hb   = (u16*)  alloc((size_t)kN * kC * 2);
    u16*   bnb  = (u16*)  alloc((size_t)kN * kC * 2);
    float* pooled = (float*)alloc((size_t)kB * kC * 4);
    float* counts = (float*)alloc(kB * 4);
    float* cf     = (float*)alloc((size_t)kB * kC * 4);

    auto cvtT = [&](const float* W, u16* Wt, int K, int Nn, int Kp) {
        int tot = Nn * Kp;
        k_cvt_transpose<<<(tot + 255) / 256, 256, 0, stream>>>(W, Wt, K, Nn, Kp, tot);
    };
    auto gemm = [&](const u16* Aa, const u16* Bt, const float* bias, const float* resid,
                    float* oF, u16* oB, int M, int K, int Nc, int act) {
        dim3 g(M / 128, Nc / 64);
        k_gemm<<<g, 256, 0, stream>>>(Aa, Bt, bias, resid, oF, oB, M, K, Nc, act);
    };
    auto g3 = [&](const u16* Kb, const u16* Eb, const u16* Bt, const float* bias, float* oF) {
        dim3 g(kE / 128, kC / 64);
        k_gemm_g3<<<g, 256, 0, stream>>>(Kb, Eb, ei, Bt, bias, oF, kE);
    };

    // ---- weight prep (bf16, N-major so B fragments are contiguous-K) ----
    cvtT(atom_W1, tW_a1, 92, 256, 96);
    cvtT(atom_W2, tW_a2, 256, 256, 256);
    cvtT(edge_W, tW_ed, 256, 256, 256);
    cvtT(pdd_W, tW_pdd, 51, 256, 64);
    for (int i = 0; i < 3; ++i) {
        cvtT(Wq + (size_t)i * kC * kC, tW_q + (size_t)i * kC * kC, 256, 256, 256);
        cvtT(Wk + (size_t)i * kC * kC, tW_k + (size_t)i * kC * kC, 256, 256, 256);
        cvtT(Wv + (size_t)i * kC * kC, tW_v + (size_t)i * kC * kC, 256, 256, 256);
        cvtT(We + (size_t)i * kC * kC, tW_e + (size_t)i * kC * kC, 256, 256, 256);
        cvtT(Wm + (size_t)i * kC * kC, tW_m + (size_t)i * kC * kC, 256, 256, 256);
        cvtT(Wku + (size_t)i * 768 * kC, tW_ku + (size_t)i * kC * 768, 768, 256, 768);
        cvtT(Wmu + (size_t)i * 768 * kC, tW_mu + (size_t)i * kC * 768, 768, 256, 768);
    }
    for (int j = 0; j < 2; ++j) {
        cvtT(pW1 + (size_t)j * kC * 512, tW_p1 + (size_t)j * 512 * kC, 256, 512, 256);
        cvtT(pW2 + (size_t)j * kC * kC, tW_p2 + (size_t)j * kC * kC, 256, 256, 256);
        cvtT(pW3 + (size_t)j * kC * kC, tW_p3 + (size_t)j * kC * kC, 256, 256, 256);
    }

    // ---- embeddings ----
    k_pad_cvt<<<(kN * 96 + 255) / 256, 256, 0, stream>>>(node, nodeb, 92, 96, kN * 96);
    k_pad_cvt<<<(kN * 64 + 255) / 256, 256, 0, stream>>>(pdd, pddb, 51, 64, kN * 64);
    gemm(nodeb, tW_a1, atom_b1, nullptr, nullptr, h1b, kN, 96, kC, 1);  // silu
    gemm(h1b, tW_a2, atom_b2, nullptr, xf, xb, kN, kC, kC, 0);          // x
    gemm(pddb, tW_pdd, pdd_b, nullptr, pf, nullptr, kN, 64, kC, 0);     // p (adj)
    k_rbf<<<kE, 256, 0, stream>>>(edge_attr, ebuf);                      // rbf (bf16)
    gemm(ebuf, tW_ed, edge_b, nullptr, nullptr, einb, kE, kC, kC, 1);   // e_in = silu(...)

    // ---- 3 conv layers, 2 pddc layers interleaved ----
    for (int L = 0; L < 3; ++L) {
        gemm(xb, tW_q + (size_t)L * kC * kC, bq + L * kC, nullptr, qf, nullptr, kN, kC, kC, 0);
        gemm(xb, tW_k + (size_t)L * kC * kC, bk + L * kC, nullptr, nullptr, kbuf, kN, kC, kC, 0);
        gemm(xb, tW_v + (size_t)L * kC * kC, bv + L * kC, nullptr, nullptr, vbuf, kN, kC, kC, 0);
        gemm(einb, tW_e + (size_t)L * kC * kC, be + L * kC, nullptr, nullptr, ebuf, kE, kC, kC, 0);
        g3(kbuf, ebuf, tW_ku + (size_t)L * kC * 768, bku + L * kC, keyE);
        g3(vbuf, ebuf, tW_mu + (size_t)L * kC * 768, bmu + L * kC, msgR);
        k_alpha_gate<<<kE, 256, 0, stream>>>(qf, keyE, msgR, ei,
                                             ln_a_g + L * kC, ln_a_b + L * kC, ebuf);
        gemm(ebuf, tW_m + (size_t)L * kC * kC, bm + L * kC, nullptr, keyE, nullptr, kE, kC, kC, 0);
        hipMemsetAsync(agg, 0, (size_t)kN * kC * 4, stream);
        k_ln_scatter<<<kE, 256, 0, stream>>>(keyE, ei, ln_m_g + L * kC, ln_m_b + L * kC, agg);
        hipMemsetAsync(cs, 0, kC * 4, stream);
        hipMemsetAsync(csq, 0, kC * 4, stream);
        k_colstats<<<kN / 256, 256, 0, stream>>>(agg, cs, csq);
        k_bn_softplus<<<kN * kC / 256, 256, 0, stream>>>(xf, agg, cs, csq,
                                                         bn_g + L * kC, bn_b + L * kC, xf, xb);
        if (L < 2) {
            int j = L;
            k_add<<<kN * kC / 256, 256, 0, stream>>>(pf, xf);
            hipMemsetAsync(cs, 0, kC * 4, stream);
            hipMemsetAsync(csq, 0, kC * 4, stream);
            k_colstats<<<kN / 256, 256, 0, stream>>>(pf, cs, csq);
            k_bn_cvt<<<kN * kC / 256, 256, 0, stream>>>(pf, cs, csq,
                                                        pbn_g + j * kC, pbn_b + j * kC, bnb);
            gemm(bnb, tW_p1 + (size_t)j * 512 * kC, pb1 + j * 512, nullptr, a2, nullptr, kN, kC, 512, 0);
            k_split<<<kN * kC / 256, 256, 0, stream>>>(a2, x1b, x2f);
            gemm(x1b, tW_p2 + (size_t)j * kC * kC, pb2 + j * kC, nullptr, x1o, nullptr, kN, kC, kC, 0);
            k_mul_cvt<<<kN * kC / 256, 256, 0, stream>>>(x1o, x2f, hb);
            gemm(hb, tW_p3 + (size_t)j * kC * kC, pb3 + j * kC, xf, xf, xb, kN, kC, kC, 0);
        }
    }

    // ---- pooling + head ----
    hipMemsetAsync(pooled, 0, (size_t)kB * kC * 4, stream);
    hipMemsetAsync(counts, 0, kB * 4, stream);
    k_count<<<kN / 256, 256, 0, stream>>>(batch, counts);
    k_pool<<<kN * kC / 256, 256, 0, stream>>>(xf, batch, pooled);
    k_poolfin<<<kB * kC / 256, 256, 0, stream>>>(pooled, counts);
    k_head1<<<kB, 256, 0, stream>>>(pooled, fc_W, fc_b, cf);
    k_head2<<<kB, 256, 0, stream>>>(cf, out_W, out_b, outp);
}
